// SSINF3Layer_19198503813128
// MI455X (gfx1250) — matrix operand order repaired and checked by reading, not compiled, not run on hardware
//
#include <hip/hip_runtime.h>
#include <stdint.h>

typedef unsigned short u16;
typedef __attribute__((ext_vector_type(16))) __bf16 v16bf;
typedef __attribute__((ext_vector_type(8)))  float  v8f;
typedef __attribute__((ext_vector_type(4)))  unsigned int u32x4;

#define T_TOK 8192
#define DIM   1024
#define NSUB  64
#define NR    8
#define NP    3
#define KTOP  6
#define HL    2574
#define HLP   2688   // padded to multiple of 128 (block N tile)
#define HG    256
#define SRCOL 512    // S*R

__device__ __forceinline__ u16 f2bf(float f) {
    unsigned u = __float_as_uint(f);
    unsigned r = (u + 0x7FFFu + ((u >> 16) & 1u)) >> 16;  // round-to-nearest-even
    return (u16)r;
}
__device__ __forceinline__ float gelu_t(float x) {
    return 0.5f * x * (1.0f + tanhf(0.7978845608028654f * (x + 0.044715f * x * x * x)));
}

union Frag { v16bf v; u32x4 q[2]; };

#define M_PLAIN      0
#define M_BIASF32    1
#define M_GELU_BF16  2
#define M_GATED_BF16 3
#define M_FINAL      4

// C[M,N] = A[M,K](bf16 row-major) * B^T[N,K](bf16).
// Per-wave tile: (RM*16) x 64. Block: 8 waves = 4(M) x 2(N) -> (RM*64) x 128.
// K must be a multiple of 64 (two-deep software pipeline over 32-wide K steps).
// A 16x32 / B 32x16 bf16 fragments use contiguous k = 16*half + j in both; f32 C/D 16x16 per CDNA5 ISA 7.12.2.
template<int MODE, int RM>
__global__ __launch_bounds__(256) void gemm_bf16_wmma(
    const u16* __restrict__ A, const u16* __restrict__ B1, const u16* __restrict__ B2,
    const float* bias1, const float* bias2, const float* addend,
    float* outF, u16* outB, int N, int K, int biasN)
{
    const int tid  = threadIdx.x;
    const int wave = tid >> 5, lane = tid & 31;
    const int nl   = lane & 15, half = lane >> 4;
    const int row0 = blockIdx.y * (RM * 64) + (wave & 3) * (RM * 16);
    const int col0 = blockIdx.x * 128 + (wave >> 2) * 64;

    v8f z = {0.f,0.f,0.f,0.f,0.f,0.f,0.f,0.f};
    v8f acc[4 * RM], acc2[4 * RM];
#pragma unroll
    for (int i = 0; i < 4 * RM; ++i) { acc[i] = z; acc2[i] = z; }

    Frag aF[2][RM], b1F[2][4], b2F[2][4];

    auto loadStep = [&](int buf, int kk) {
#pragma unroll
        for (int m = 0; m < RM; ++m) {
            // lane (row=nl, half): k = half*16 + j  (contiguous 16 elements)
            const u16* ar = A + (size_t)(row0 + m * 16 + nl) * K + kk;
            aF[buf][m].q[0] = *(const u32x4*)(ar + half * 16);
            aF[buf][m].q[1] = *(const u32x4*)(ar + half * 16 + 8);
        }
#pragma unroll
        for (int s = 0; s < 4; ++s) {
            // lane (col=nl, half): k = half*16 + j  (contiguous 16 elements)
            const u16* br = B1 + (size_t)(col0 + s * 16 + nl) * K + kk + half * 16;
            b1F[buf][s].q[0] = ((const u32x4*)br)[0];
            b1F[buf][s].q[1] = ((const u32x4*)br)[1];
            if (MODE == M_GATED_BF16) {
                const u16* gr = B2 + (size_t)(col0 + s * 16 + nl) * K + kk + half * 16;
                b2F[buf][s].q[0] = ((const u32x4*)gr)[0];
                b2F[buf][s].q[1] = ((const u32x4*)gr)[1];
            }
        }
    };
    auto mmaStep = [&](int buf) {
#pragma unroll
        for (int s = 0; s < 4; ++s)
#pragma unroll
            for (int m = 0; m < RM; ++m) {
                acc[s * RM + m] = __builtin_amdgcn_wmma_f32_16x16x32_bf16(
                    false, aF[buf][m].v, false, b1F[buf][s].v, (short)0,
                    acc[s * RM + m], false, false);
                if (MODE == M_GATED_BF16)
                    acc2[s * RM + m] = __builtin_amdgcn_wmma_f32_16x16x32_bf16(
                        false, aF[buf][m].v, false, b2F[buf][s].v, (short)0,
                        acc2[s * RM + m], false, false);
            }
    };

    // two-deep software pipeline: loads for the next 32-step are in flight
    // while the WMMAs for the current step execute.
    loadStep(0, 0);
    for (int kk = 0; kk < K; kk += 64) {
        loadStep(1, kk + 32);
        mmaStep(0);
        if (kk + 64 < K) loadStep(0, kk + 64);
        mmaStep(1);
    }

#pragma unroll
    for (int s = 0; s < 4; ++s) {
        int c = col0 + s * 16 + nl;
#pragma unroll
        for (int m = 0; m < RM; ++m) {
#pragma unroll
            for (int i = 0; i < 8; ++i) {
                int r = row0 + m * 16 + half * 8 + i;   // C/D: VGPR i -> M = 8*half + i
                size_t o = (size_t)r * N + c;
                float v = acc[s * RM + m][i];
                if      (MODE == M_PLAIN)     outF[o] = v;
                else if (MODE == M_BIASF32)   outF[o] = v + bias1[c];
                else if (MODE == M_GELU_BF16) outB[o] = f2bf(gelu_t(v + bias1[c]));
                else if (MODE == M_GATED_BF16) {
                    float b1v = (c < biasN) ? bias1[c] : 0.f;
                    float b2v = (c < biasN) ? bias2[c] : 0.f;
                    outB[o] = f2bf(gelu_t(v + b1v) * (acc2[s * RM + m][i] + b2v));
                } else { // M_FINAL
                    outF[o] = v + bias1[c] + addend[o];
                }
            }
        }
    }
}

__global__ void cast_f32_to_bf16(const float* __restrict__ in, u16* __restrict__ out, size_t n) {
    size_t i = (size_t)blockIdx.x * blockDim.x + threadIdx.x;
    size_t stride = (size_t)gridDim.x * blockDim.x;
    for (; i < n; i += stride) out[i] = f2bf(in[i]);
}

// dst[n*Kpad + k] = (k<Ksrc && n<Nsrc) ? src[k*Nsrc + n] : 0   (src is [Ksrc, Nsrc] f32)
__global__ void transpose_cast_bf16(const float* __restrict__ src, u16* __restrict__ dst,
                                    int Ksrc, int Nsrc, int Kpad, int Npad) {
    long total = (long)Kpad * Npad;
    long i = (long)blockIdx.x * blockDim.x + threadIdx.x;
    if (i >= total) return;
    int k = (int)(i % Kpad);
    int n = (int)(i / Kpad);
    float v = (k < Ksrc && n < Nsrc) ? src[(size_t)k * Nsrc + n] : 0.0f;
    dst[i] = f2bf(v);
}

// One thread per token: dist^2 over 64 subspaces, top-6, softmax(T=2), Hermite splanifold.
__global__ __launch_bounds__(256) void routing_kernel(
    const float* __restrict__ coords, const float* __restrict__ centers,
    const float* __restrict__ sigma, const float* __restrict__ extrap,
    const float* __restrict__ a0_, const float* __restrict__ a1_,
    const float* __restrict__ b0_, const float* __restrict__ b1_,
    const float* __restrict__ pt0_, const float* __restrict__ pt1_,
    const float* __restrict__ bt0_, const float* __restrict__ bt1_,
    float* __restrict__ wvals, int* __restrict__ idxout)
{
    __shared__ float sc[SRCOL];
    for (int i = threadIdx.x; i < SRCOL; i += blockDim.x) sc[i] = centers[i];
    __syncthreads();

    int t = blockIdx.x * blockDim.x + threadIdx.x;
    if (t >= T_TOK) return;
    const float* cr = coords + (size_t)t * SRCOL;

    float bd[KTOP]; int bi[KTOP];
#pragma unroll
    for (int k = 0; k < KTOP; ++k) { bd[k] = 3.4e38f; bi[k] = 0; }
    for (int s = 0; s < NSUB; ++s) {
        float d2 = 0.f;
#pragma unroll
        for (int r = 0; r < NR; ++r) {
            float diff = cr[s * NR + r] - sc[s * NR + r];
            d2 += diff * diff;
        }
        if (d2 < bd[KTOP - 1]) {      // strict < keeps lower index first on ties (matches top_k)
            int j = KTOP - 1;
            while (j > 0 && d2 < bd[j - 1]) { bd[j] = bd[j - 1]; bi[j] = bi[j - 1]; --j; }
            bd[j] = d2; bi[j] = s;
        }
    }
    // softmax over scores = -d2, temperature 2.0; bd sorted ascending -> bd[0] is max score
    float mx = -bd[0];
    float e[KTOP], sum = 0.f;
#pragma unroll
    for (int k = 0; k < KTOP; ++k) { e[k] = expf((-bd[k] - mx) * 0.5f); sum += e[k]; }
    float inv = 1.0f / sum;

#pragma unroll
    for (int k = 0; k < KTOP; ++k) {
        int s = bi[k];
        float rw  = e[k] * inv;
        float sig = sigma[s], ex = extrap[s];
        float u[NR], dl[NR], ww[NR];
        float su = 0.f, umax = 0.f;
#pragma unroll
        for (int r = 0; r < NR; ++r) {
            u[r] = cr[s * NR + r] * (1.f + 2.f * ex) - ex;
            su += u[r];
            umax = fmaxf(umax, fabsf(u[r]));
        }
        float tm  = su * (1.0f / NR);
        float eps = fmaxf(umax * 1e-3f, 1e-6f);
        bool  fb  = fabsf(su) < eps;
        float safe = fb ? (su >= 0.f ? eps : -eps) : su;
#pragma unroll
        for (int r = 0; r < NR; ++r) {
            dl[r] = u[r] - tm;
            ww[r] = fb ? (1.0f / NR) : u[r] / safe;
        }
        float t2 = tm * tm, t3 = t2 * tm;
        float h00 = 2.f*t3 - 3.f*t2 + 1.f, h10 = t3 - 2.f*t2 + tm;
        float h01 = -2.f*t3 + 3.f*t2,      h11 = t3 - t2;
#pragma unroll
        for (int p = 0; p < NP; ++p) {
            float a0 = a0_[s * NP + p], a1 = a1_[s * NP + p];
            float v0 = 0.f, v1 = 0.f, d0 = 0.f, d1 = 0.f, tg0 = 0.f, tg1 = 0.f;
#pragma unroll
            for (int r = 0; r < NR; ++r) {
                int o = (s * NR + r) * NP + p;
                v0 += ww[r] * (pt0_[o] - a0);
                v1 += ww[r] * (pt1_[o] - a1);
                float bb0 = b0_[o], bb1 = b1_[o];
                d0  += dl[r] * bb0;
                d1  += dl[r] * bb1;
                tg0 += dl[r] * (bt0_[o] - bb0);
                tg1 += dl[r] * (bt1_[o] - bb1);
            }
            v0 *= sig; v1 *= sig; tg0 *= sig; tg1 *= sig;
            float val = h00*a0 + h01*a1 + h10*v0 + h11*v1
                      + h00*d0 + h01*d1 + h10*tg0 + h11*tg1;
            wvals[((size_t)t * KTOP + k) * NP + p] = val * rw;
        }
        idxout[(size_t)t * KTOP + k] = s;
    }
}

// One block per token: local_in[t,d] = sum_{k,p} wvals[t,k,p] * output_basis[idx[t,k],p,d]
__global__ __launch_bounds__(256) void contrib_kernel(
    const float* __restrict__ wvals, const int* __restrict__ idx,
    const float* __restrict__ ob, u16* __restrict__ lin)
{
    __shared__ float swv[KTOP * NP];
    __shared__ int   sid[KTOP];
    int t = blockIdx.x;
    if (threadIdx.x < KTOP * NP) swv[threadIdx.x] = wvals[(size_t)t * KTOP * NP + threadIdx.x];
    if (threadIdx.x < KTOP)      sid[threadIdx.x] = idx[(size_t)t * KTOP + threadIdx.x];
    __syncthreads();
    for (int d = threadIdx.x; d < DIM; d += blockDim.x) {
        float acc = 0.f;
#pragma unroll
        for (int k = 0; k < KTOP; ++k) {
            const float* base = ob + (size_t)sid[k] * NP * DIM + d;
            acc += swv[k*NP+0] * base[0] + swv[k*NP+1] * base[DIM] + swv[k*NP+2] * base[2*DIM];
        }
        lin[(size_t)t * DIM + d] = f2bf(acc);
    }
}

extern "C" void kernel_launch(void* const* d_in, const int* in_sizes, int n_in,
                              void* d_out, int out_size, void* d_ws, size_t ws_size,
                              hipStream_t stream) {
    const float* x      = (const float*)d_in[0];
    const float* ibm    = (const float*)d_in[1];
    const float* ctrs   = (const float*)d_in[2];
    const float* ob     = (const float*)d_in[3];
    const float* a0     = (const float*)d_in[4];
    const float* a1     = (const float*)d_in[5];
    const float* b0     = (const float*)d_in[6];
    const float* b1     = (const float*)d_in[7];
    const float* pt0    = (const float*)d_in[8];
    const float* pt1    = (const float*)d_in[9];
    const float* bt0    = (const float*)d_in[10];
    const float* bt1    = (const float*)d_in[11];
    const float* sigma  = (const float*)d_in[12];
    const float* extrap = (const float*)d_in[13];
    const float* lwi    = (const float*)d_in[14];
    const float* lbi    = (const float*)d_in[15];
    const float* lwo    = (const float*)d_in[16];
    const float* lbo    = (const float*)d_in[17];
    const float* lwg    = (const float*)d_in[18];
    const float* lbg    = (const float*)d_in[19];
    const float* giw    = (const float*)d_in[20];
    const float* gib    = (const float*)d_in[21];
    const float* gow    = (const float*)d_in[22];
    const float* gob    = (const float*)d_in[23];
    float* out = (float*)d_out;

    char* wsb = (char*)d_ws;
    size_t off = 0;
    auto alloc = [&](size_t bytes) -> void* {
        off = (off + 255) & ~(size_t)255;
        void* p = wsb + off;
        off += bytes;
        return p;
    };
    u16*   ws_x    = (u16*)alloc((size_t)T_TOK * DIM * 2);
    u16*   ws_ib   = (u16*)alloc((size_t)SRCOL * DIM * 2);
    u16*   ws_giw  = (u16*)alloc((size_t)HG * DIM * 2);
    u16*   ws_gow  = (u16*)alloc((size_t)DIM * HG * 2);
    u16*   ws_lwi  = (u16*)alloc((size_t)HLP * DIM * 2);
    u16*   ws_lwg  = (u16*)alloc((size_t)HLP * DIM * 2);
    u16*   ws_lwo  = (u16*)alloc((size_t)DIM * HLP * 2);
    u16*   ws_gh   = (u16*)alloc((size_t)T_TOK * HG * 2);
    float* ws_co   = (float*)alloc((size_t)T_TOK * SRCOL * 4);
    float* ws_wv   = (float*)alloc((size_t)T_TOK * KTOP * NP * 4);
    int*   ws_idx  = (int*)alloc((size_t)T_TOK * KTOP * 4);
    u16*   ws_lin  = (u16*)alloc((size_t)T_TOK * DIM * 2);
    u16*   ws_h    = (u16*)alloc((size_t)T_TOK * HLP * 2);
    (void)in_sizes; (void)n_in; (void)out_size; (void)ws_size;

    // --- precision conversions / transposes into WMMA-friendly B^T layouts ---
    cast_f32_to_bf16<<<2048, 256, 0, stream>>>(x,   ws_x,   (size_t)T_TOK * DIM);
    cast_f32_to_bf16<<<256,  256, 0, stream>>>(giw, ws_giw, (size_t)HG * DIM);    // already [N,K]
    cast_f32_to_bf16<<<256,  256, 0, stream>>>(gow, ws_gow, (size_t)DIM * HG);    // already [N,K]
    transpose_cast_bf16<<<((long)SRCOL*DIM + 255)/256, 256, 0, stream>>>(ibm, ws_ib, DIM, SRCOL, DIM, SRCOL);
    transpose_cast_bf16<<<((long)HLP*DIM + 255)/256,  256, 0, stream>>>(lwi, ws_lwi, DIM, HL, DIM, HLP);
    transpose_cast_bf16<<<((long)HLP*DIM + 255)/256,  256, 0, stream>>>(lwg, ws_lwg, DIM, HL, DIM, HLP);
    transpose_cast_bf16<<<((long)DIM*HLP + 255)/256,  256, 0, stream>>>(lwo, ws_lwo, HL, DIM, HLP, DIM);

    // --- global MLP: g_h = gelu(x @ giw^T + gib);  g_out = g_h @ gow^T + gob -> d_out ---
    gemm_bf16_wmma<M_GELU_BF16, 2><<<dim3(HG/128, T_TOK/128), 256, 0, stream>>>(
        ws_x, ws_giw, nullptr, gib, nullptr, nullptr, nullptr, ws_gh, HG, DIM, HG);
    gemm_bf16_wmma<M_BIASF32, 2><<<dim3(DIM/128, T_TOK/128), 256, 0, stream>>>(
        ws_gh, ws_gow, nullptr, gob, nullptr, nullptr, out, nullptr, DIM, HG, DIM);

    // --- routing coords = x @ input_basis_matrix ---
    gemm_bf16_wmma<M_PLAIN, 2><<<dim3(SRCOL/128, T_TOK/128), 256, 0, stream>>>(
        ws_x, ws_ib, nullptr, nullptr, nullptr, nullptr, ws_co, nullptr, SRCOL, DIM, 0);

    // --- top-k routing + splanifold, then gather through output basis ---
    routing_kernel<<<T_TOK/256, 256, 0, stream>>>(
        ws_co, ctrs, sigma, extrap, a0, a1, b0, b1, pt0, pt1, bt0, bt1, ws_wv, ws_idx);
    contrib_kernel<<<T_TOK, 256, 0, stream>>>(ws_wv, ws_idx, ob, ws_lin);

    // --- gated local MLP (fused dual-B GEMM; RM=1 to stay under 256 VGPRs) ---
    gemm_bf16_wmma<M_GATED_BF16, 1><<<dim3(HLP/128, T_TOK/64), 256, 0, stream>>>(
        ws_lin, ws_lwi, ws_lwg, lbi, lbg, nullptr, nullptr, ws_h, HLP, DIM, HL);

    // --- local_out = h @ lwo^T + lbo + g_out -> d_out ---
    gemm_bf16_wmma<M_FINAL, 2><<<dim3(DIM/128, T_TOK/128), 256, 0, stream>>>(
        ws_h, ws_lwo, nullptr, lbo, nullptr, out, out, nullptr, DIM, HLP, DIM);
}
